// FrequencyBranch_42494406427068
// MI455X (gfx1250) — matrix operand order repaired and checked by reading, not compiled, not run on hardware
//
#include <hip/hip_runtime.h>
#include <hip/hip_bf16.h>

typedef __attribute__((ext_vector_type(16))) _Float16 v16h;
typedef __attribute__((ext_vector_type(8)))  float    v8f;

#define BB    32
#define CC    256
#define HH    56
#define WW    56
#define MM    16
#define MIXX  4
#define KK    9        // num_fourier_basis + 1
#define HIDD  16
#define WWORK 62       // (32-1)*2
#define WFQ   32       // rfft bins of 62
#define PH    14       // 56/4
#define PWF   8        // 32/4
#define HWP   (HH*WW)  // 3136
#define EPSV  1e-5f
#define PI_F  3.14159265358979323846f

// ---------------------------------------------------------------- DFT tables
__global__ void k_tables(float2* __restrict__ tWF, float2* __restrict__ tH,
                         float2* __restrict__ tWI) {
  int t = threadIdx.x;
  const float invW = rsqrtf((float)WWORK);
  const float invH = rsqrtf((float)HH);
  // forward real->complex DFT along W (ortho): X[f] = sum_w x[w]*(cos - i sin)*invW
  for (int i = t; i < WWORK * WFQ; i += blockDim.x) {
    int w = i / WFQ, f = i % WFQ;
    float ang = 2.0f * PI_F * (float)(f * w) / (float)WWORK;
    tWF[i] = make_float2(cosf(ang) * invW, sinf(ang) * invW);
  }
  // H DFT twiddles (shared by forward/backward, sign handled in kernels)
  for (int i = t; i < HH * HH; i += blockDim.x) {
    int a = i / HH, b = i % HH;
    float ang = 2.0f * PI_F * (float)(a * b) / (float)HH;
    tH[i] = make_float2(cosf(ang) * invH, sinf(ang) * invH);
  }
  // inverse rDFT along W with hermitian fold: y[w] = sum_f (Xr*c + Xi*s)
  for (int i = t; i < WFQ * WWORK; i += blockDim.x) {
    int f = i / WWORK, w = i % WWORK;
    float alpha = (f == 0 || f == WFQ - 1) ? 1.0f : 2.0f;
    float ang = 2.0f * PI_F * (float)(f * w) / (float)WWORK;
    tWI[i] = make_float2(alpha * cosf(ang) * invW, -alpha * sinf(ang) * invW);
  }
}

// ---------------------------------------------------------------- GAP
__global__ void k_gap(const float* __restrict__ x, float* __restrict__ ctx) {
  __shared__ float red[256];
  int bc = blockIdx.x;
  const float* p = x + (size_t)bc * HWP;
  float s = 0.f;
  for (int i = threadIdx.x; i < HWP; i += 256) s += p[i];
  red[threadIdx.x] = s;
  __syncthreads();
  for (int o = 128; o > 0; o >>= 1) {
    if (threadIdx.x < o) red[threadIdx.x] += red[threadIdx.x + o];
    __syncthreads();
  }
  if (threadIdx.x == 0) ctx[bc] = red[0] * (1.0f / (float)HWP);
}

// ---------------------------------------------------------------- fc1+LN+ReLU
__global__ void k_head(const float* __restrict__ ctx, const float* __restrict__ fc1_w,
                       const float* __restrict__ fc1_b, const float* __restrict__ ln_g,
                       const float* __restrict__ ln_b, float* __restrict__ hdn) {
  __shared__ float sh[BB][HIDD];
  __shared__ float smu[BB], srv[BB];
  int t = threadIdx.x;           // 512 threads = B*HID
  int b = t / HIDD, j = t % HIDD;
  const float* cr = ctx + b * CC;
  const float* wr = fc1_w + j * CC;
  float raw = fc1_b[j];
  for (int c = 0; c < CC; ++c) raw += cr[c] * wr[c];
  sh[b][j] = raw;
  __syncthreads();
  if (t < BB) {
    float m = 0.f;
    for (int q = 0; q < HIDD; ++q) m += sh[t][q];
    m *= (1.0f / HIDD);
    float v = 0.f;
    for (int q = 0; q < HIDD; ++q) { float d = sh[t][q] - m; v += d * d; }
    v *= (1.0f / HIDD);
    smu[t] = m; srv[t] = rsqrtf(v + EPSV);
  }
  __syncthreads();
  float val = (sh[b][j] - smu[b]) * srv[b] * ln_g[j] + ln_b[j];
  hdn[b * HIDD + j] = fmaxf(val, 0.f);
}

// ---------------------------------------------------------------- fourier head + softmax
__global__ void k_coeffs(const float* __restrict__ hdn, const float* __restrict__ fh_w,
                         const float* __restrict__ fh_b, float* __restrict__ coeffs) {
  int idx = blockIdx.x * blockDim.x + threadIdx.x;  // B*C*M workers
  if (idx >= BB * CC * MM) return;
  int b = idx / (CC * MM);
  int cm = idx % (CC * MM);      // c*M + m
  float h[HIDD];
  const float* hr = hdn + b * HIDD;
  #pragma unroll
  for (int j = 0; j < HIDD; ++j) h[j] = hr[j];
  float att[KK];
  float mx = -1e30f;
  for (int k = 0; k < KK; ++k) {
    int row = cm * KK + k;       // (c*M+m)*K + k
    const float* wr = fh_w + (size_t)row * HIDD;
    float a = fh_b[row];
    #pragma unroll
    for (int j = 0; j < HIDD; ++j) a += h[j] * wr[j];
    att[k] = a; mx = fmaxf(mx, a);
  }
  float s = 0.f;
  for (int k = 0; k < KK; ++k) { att[k] = __expf(att[k] - mx); s += att[k]; }
  float inv = 1.f / s;
  float* out = coeffs + (size_t)idx * KK;
  for (int k = 0; k < KK; ++k) out[k] = att[k] * inv;
}

// ------------------------------------------- bilinear 56->62 fused with W rDFT
__global__ void k_fwdW(const float* __restrict__ x, const float2* __restrict__ tWF,
                       float2* __restrict__ C1) {
  __shared__ float work[HH * WWORK];
  int bc = blockIdx.x;
  const float* img = x + (size_t)bc * HWP;
  for (int i = threadIdx.x; i < HH * WWORK; i += blockDim.x) {
    int h = i / WWORK, w2 = i % WWORK;
    float src = (w2 + 0.5f) * ((float)WW / (float)WWORK) - 0.5f;
    src = fminf(fmaxf(src, 0.f), (float)(WW - 1));
    int w0 = (int)src;
    int w1 = min(w0 + 1, WW - 1);
    float fr = src - (float)w0;
    work[i] = img[h * WW + w0] * (1.f - fr) + img[h * WW + w1] * fr;
  }
  __syncthreads();
  float2* out = C1 + (size_t)bc * HH * WFQ;
  for (int i = threadIdx.x; i < HH * WFQ; i += blockDim.x) {
    int h = i / WFQ, f = i % WFQ;
    const float* row = work + h * WWORK;
    float re = 0.f, im = 0.f;
    for (int w = 0; w < WWORK; ++w) {
      float2 cs = tWF[w * WFQ + f];
      re += row[w] * cs.x;
      im -= row[w] * cs.y;
    }
    out[i] = make_float2(re, im);
  }
}

// ------------------------------- H DFT fused with mixture spectral modulation
__global__ void k_fftH_mod(const float2* __restrict__ C1, const float2* __restrict__ tH,
                           const float* __restrict__ coeffs,
                           const float* __restrict__ basis_r,
                           const float* __restrict__ basis_i,
                           float2* __restrict__ C2) {
  __shared__ float2 sIn[HH * WFQ];
  int bc = blockIdx.x;
  const float2* in = C1 + (size_t)bc * HH * WFQ;
  for (int i = threadIdx.x; i < HH * WFQ; i += blockDim.x) sIn[i] = in[i];
  __syncthreads();
  float2* out = C2 + (size_t)bc * HH * WFQ;
  for (int i = threadIdx.x; i < HH * WFQ; i += blockDim.x) {
    int hp = i / WFQ, f = i % WFQ;
    const float2* tr = tH + hp * HH;
    float re = 0.f, im = 0.f;
    for (int h = 0; h < HH; ++h) {          // forward: * (cos - i sin)
      float2 cs = tr[h];
      float2 a = sIn[h * WFQ + f];
      re += a.x * cs.x + a.y * cs.y;
      im += a.y * cs.x - a.x * cs.y;
    }
    int m = (hp / PH) * MIXX + (f / PWF);   // mixture block id
    const float* cf = coeffs + ((size_t)bc * MM + m) * KK;
    float kr = 0.f, ki = 0.f;
    #pragma unroll
    for (int k = 0; k < KK; ++k) {
      float c = cf[k];
      kr += c * basis_r[(k * HH + hp) * WFQ + f];
      ki += c * basis_i[(k * HH + hp) * WFQ + f];
    }
    out[i] = make_float2(re * kr - im * ki, re * ki + im * kr);
  }
}

// ---------------------------------------------------------------- inverse H DFT
__global__ void k_ifftH(const float2* __restrict__ C2, const float2* __restrict__ tH,
                        float2* __restrict__ C1) {
  __shared__ float2 sIn[HH * WFQ];
  int bc = blockIdx.x;
  const float2* in = C2 + (size_t)bc * HH * WFQ;
  for (int i = threadIdx.x; i < HH * WFQ; i += blockDim.x) sIn[i] = in[i];
  __syncthreads();
  float2* out = C1 + (size_t)bc * HH * WFQ;
  for (int i = threadIdx.x; i < HH * WFQ; i += blockDim.x) {
    int h = i / WFQ, f = i % WFQ;
    const float2* tr = tH + h * HH;
    float re = 0.f, im = 0.f;
    for (int hp = 0; hp < HH; ++hp) {       // inverse: * (cos + i sin)
      float2 cs = tr[hp];
      float2 a = sIn[hp * WFQ + f];
      re += a.x * cs.x - a.y * cs.y;
      im += a.y * cs.x + a.x * cs.y;
    }
    out[i] = make_float2(re, im);
  }
}

// -------------------------- inverse W rDFT fused with bilinear 62->56 resize
__global__ void k_invW(const float2* __restrict__ C1, const float2* __restrict__ tWI,
                       float* __restrict__ y) {
  __shared__ float2 sf[HH * WFQ];
  __shared__ float ywork[HH * WWORK];
  int bc = blockIdx.x;
  const float2* in = C1 + (size_t)bc * HH * WFQ;
  for (int i = threadIdx.x; i < HH * WFQ; i += blockDim.x) sf[i] = in[i];
  __syncthreads();
  for (int i = threadIdx.x; i < HH * WWORK; i += blockDim.x) {
    int h = i / WWORK, w = i % WWORK;
    float acc = 0.f;
    for (int f = 0; f < WFQ; ++f) {
      float2 cs = tWI[f * WWORK + w];
      float2 a = sf[h * WFQ + f];
      acc += a.x * cs.x + a.y * cs.y;
    }
    ywork[i] = acc;
  }
  __syncthreads();
  float* out = y + (size_t)bc * HWP;
  for (int i = threadIdx.x; i < HWP; i += blockDim.x) {
    int h = i / WW, w2 = i % WW;
    float src = (w2 + 0.5f) * ((float)WWORK / (float)WW) - 0.5f;
    src = fminf(fmaxf(src, 0.f), (float)(WWORK - 1));
    int w0 = (int)src;
    int w1 = min(w0 + 1, WWORK - 1);
    float fr = src - (float)w0;
    out[i] = ywork[h * WWORK + w0] * (1.f - fr) + ywork[h * WWORK + w1] * fr;
  }
}

// -------------------------------- 1x1 conv as WMMA GEMM: Out = W(256x256) x Y
// Block tile 32(o) x 64(p), 4 waves; each wave: 16x32 via two v_wmma accumulators
// sharing one A fragment. Tiles double-buffered in LDS via
// GLOBAL_LOAD_ASYNC_TO_LDS_B128 (ASYNCcnt): tile k+1 is in flight while tile k
// is consumed (s_wait_asynccnt 6 waits only for the 6 older per-wave async ops,
// which complete in order per ISA async ordering). Fully unrolled K loop: 16 WMMAs.
__global__ void k_conv_wmma(const float* __restrict__ y, const float* __restrict__ w,
                            float* __restrict__ y2) {
  __shared__ float sW[2][32 * 32];   // [buf][o][k] tiles, 8KB
  __shared__ float sY[2][32 * 64];   // [buf][k][p] tiles, 16KB
  int t = threadIdx.x;               // 128 threads = 4 waves
  int lane = t & 31;
  int wv = t >> 5;
  int wo = wv >> 1;                  // wave o-half {0,1}
  int wp = wv & 1;                   // wave p-half {0,1}, 32 cols each
  int blk = blockIdx.x;
  int b  = blk / (8 * 49);
  int r  = blk % (8 * 49);
  int o0 = (r / 49) * 32;
  int p0 = (r % 49) * 64;
  int m  = lane & 15;                // fragment row/col within tile
  int hi = lane >> 4;                // lane half selector
  v8f acc0 = {}, acc1 = {};
  const float* ybase = y + (size_t)b * CC * HWP;

  auto stage = [&](int kk, int buf) {
    int c0 = kk * 32;
    // W tile (32o x 32k): 256 x B128 -> 2 async ops per thread
    #pragma unroll
    for (int q0 = 0; q0 < 2; ++q0) {
      int q = q0 * 128 + t;
      int row = q >> 3, c4 = q & 7;
      unsigned lo = (unsigned)(uintptr_t)(&sW[buf][row * 32 + c4 * 4]);
      unsigned long long ga =
          (unsigned long long)(uintptr_t)(w + (size_t)(o0 + row) * CC + c0 + c4 * 4);
      asm volatile("global_load_async_to_lds_b128 %0, %1, off"
                   :: "v"(lo), "v"(ga) : "memory");
    }
    // Y tile (32k x 64p): 512 x B128 -> 4 async ops per thread
    #pragma unroll
    for (int q0 = 0; q0 < 4; ++q0) {
      int q = q0 * 128 + t;
      int row = q >> 4, c4 = q & 15;
      unsigned lo = (unsigned)(uintptr_t)(&sY[buf][row * 64 + c4 * 4]);
      unsigned long long ga =
          (unsigned long long)(uintptr_t)(ybase + (size_t)(c0 + row) * HWP + p0 + c4 * 4);
      asm volatile("global_load_async_to_lds_b128 %0, %1, off"
                   :: "v"(lo), "v"(ga) : "memory");
    }
  };

  stage(0, 0);                       // prologue: tile 0 in flight
  #pragma unroll
  for (int kk = 0; kk < 8; ++kk) {
    int buf = kk & 1;
    if (kk + 1 < 8) {
      stage(kk + 1, buf ^ 1);        // prefetch next tile into other buffer
      asm volatile("s_wait_asynccnt 6" ::: "memory");   // tile kk done, kk+1 in flight
    } else {
      asm volatile("s_wait_asynccnt 0" ::: "memory");
    }
    __syncthreads();                 // tile kk visible to all waves
    // --- fragments from LDS ---
    // A 16x32 f16: lane<16 holds K {0..15}, lane>=16 holds {16..31}
    v16h A, B0, B1;
    int kb = hi * 16;
    const float* wr = &sW[buf][(wo * 16 + m) * 32];
    #pragma unroll
    for (int e = 0; e < 8; ++e) {
      A[e]     = (_Float16)wr[kb + e];
      A[8 + e] = (_Float16)wr[kb + 8 + e];
    }
    // B 32x16 f16: lane holds column (lane%16), K = hi*16 + e
    int koff = hi * 16;
    #pragma unroll
    for (int e = 0; e < 16; ++e) {
      B0[e] = (_Float16)sY[buf][(koff + e) * 64 + wp * 32 + m];
      B1[e] = (_Float16)sY[buf][(koff + e) * 64 + wp * 32 + 16 + m];
    }
    acc0 = __builtin_amdgcn_wmma_f32_16x16x32_f16(false, A, false, B0,
                                                  (short)0, acc0, false, false);
    acc1 = __builtin_amdgcn_wmma_f32_16x16x32_f16(false, A, false, B1,
                                                  (short)0, acc1, false, false);
    __syncthreads();                 // reads done before buffer reused (dist 2)
  }
  // D: row = hi*8 + rr, col = lane%16
  float* ob = y2 + ((size_t)b * CC + o0 + wo * 16 + hi * 8) * HWP + p0 + wp * 32 + m;
  #pragma unroll
  for (int rr = 0; rr < 8; ++rr) {
    ob[(size_t)rr * HWP]      = acc0[rr];
    ob[(size_t)rr * HWP + 16] = acc1[rr];
  }
}

// ---------------------------------------------------------------- BN stats (1 block/channel)
__global__ void k_bn_stats(const float* __restrict__ y2, float* __restrict__ stats) {
  __shared__ float rs[256], rq[256];
  int o = blockIdx.x;
  float s = 0.f, q = 0.f;
  for (int i = threadIdx.x; i < BB * HWP; i += 256) {
    int b = i / HWP, p = i % HWP;
    float v = y2[((size_t)b * CC + o) * HWP + p];
    s += v; q += v * v;
  }
  rs[threadIdx.x] = s; rq[threadIdx.x] = q;
  __syncthreads();
  for (int st = 128; st > 0; st >>= 1) {
    if (threadIdx.x < st) {
      rs[threadIdx.x] += rs[threadIdx.x + st];
      rq[threadIdx.x] += rq[threadIdx.x + st];
    }
    __syncthreads();
  }
  if (threadIdx.x == 0) { stats[o * 2] = rs[0]; stats[o * 2 + 1] = rq[0]; }
}

__global__ void k_bn_norm(const float* __restrict__ y2, const float* __restrict__ stats,
                          const float* __restrict__ g, const float* __restrict__ bta,
                          float* __restrict__ out) {
  size_t i = (size_t)blockIdx.x * blockDim.x + threadIdx.x;
  if (i >= (size_t)BB * CC * HWP) return;
  int o = (int)((i / HWP) % CC);
  const float cnt = (float)(BB * HWP);
  float mean = stats[o * 2] / cnt;
  float var  = stats[o * 2 + 1] / cnt - mean * mean;
  out[i] = (y2[i] - mean) * rsqrtf(var + EPSV) * g[o] + bta[o];
}

// ---------------------------------------------------------------- launcher
extern "C" void kernel_launch(void* const* d_in, const int* in_sizes, int n_in,
                              void* d_out, int out_size, void* d_ws, size_t ws_size,
                              hipStream_t stream) {
  (void)in_sizes; (void)n_in; (void)out_size; (void)ws_size;
  const float* x       = (const float*)d_in[0];
  const float* fc1_w   = (const float*)d_in[1];
  const float* fc1_b   = (const float*)d_in[2];
  const float* ln_g    = (const float*)d_in[3];
  const float* ln_b    = (const float*)d_in[4];
  const float* fh_w    = (const float*)d_in[5];
  const float* fh_b    = (const float*)d_in[6];
  const float* basis_r = (const float*)d_in[7];
  const float* basis_i = (const float*)d_in[8];
  const float* conv_w  = (const float*)d_in[9];
  const float* bn_g    = (const float*)d_in[10];
  const float* bn_b    = (const float*)d_in[11];
  float* out = (float*)d_out;

  float* ws = (float*)d_ws;
  size_t off = 0;
  float2* tWF = (float2*)(ws + off); off += (size_t)WWORK * WFQ * 2;
  float2* tH  = (float2*)(ws + off); off += (size_t)HH * HH * 2;
  float2* tWI = (float2*)(ws + off); off += (size_t)WFQ * WWORK * 2;
  float* ctx  = ws + off; off += (size_t)BB * CC;
  float* hdn  = ws + off; off += (size_t)BB * HIDD;
  float* coef = ws + off; off += (size_t)BB * CC * MM * KK;
  float2* C1  = (float2*)(ws + off); off += (size_t)BB * CC * HH * WFQ * 2;
  float2* C2  = (float2*)(ws + off); off += (size_t)BB * CC * HH * WFQ * 2;
  float* stats = ws + off; off += (size_t)CC * 2;
  float* y2 = (float*)C2;  // C2 is dead after k_ifftH; reuse for conv output

  k_tables<<<1, 256, 0, stream>>>(tWF, tH, tWI);
  k_gap<<<BB * CC, 256, 0, stream>>>(x, ctx);
  k_head<<<1, 512, 0, stream>>>(ctx, fc1_w, fc1_b, ln_g, ln_b, hdn);
  k_coeffs<<<(BB * CC * MM) / 256, 256, 0, stream>>>(hdn, fh_w, fh_b, coef);
  k_fwdW<<<BB * CC, 256, 0, stream>>>(x, tWF, C1);
  k_fftH_mod<<<BB * CC, 256, 0, stream>>>(C1, tH, coef, basis_r, basis_i, C2);
  k_ifftH<<<BB * CC, 256, 0, stream>>>(C2, tH, C1);
  k_invW<<<BB * CC, 256, 0, stream>>>(C1, tWI, out);     // d_out holds y (pre-conv)
  k_conv_wmma<<<BB * 8 * 49, 128, 0, stream>>>(out, conv_w, y2);
  k_bn_stats<<<CC, 256, 0, stream>>>(y2, stats);
  k_bn_norm<<<(BB * CC * HWP + 255) / 256, 256, 0, stream>>>(y2, stats, bn_g, bn_b, out);
}
